// MultiHeadAttnBlock_6167573037776
// MI455X (gfx1250) — matrix operand order repaired and checked by reading, not compiled, not run on hardware
//
#include <hip/hip_runtime.h>

#define DEV __device__ __forceinline__

// ---------------- problem constants ----------------
constexpr int Bc  = 4;
constexpr int Tc  = 2048;
constexpr int Dc  = 1024;
constexpr int Hc  = 8;
constexpr int DHc = 128;
constexpr int BT  = Bc * Tc;   // 8192 rows

// ---------------- WMMA types ----------------
typedef __bf16 bf16_t;
typedef bf16_t v16bf __attribute__((ext_vector_type(16)));
typedef float  v8f   __attribute__((ext_vector_type(8)));

union Frag {
  v16bf v;
  uint4 q[2];
};

DEV unsigned short f32_to_bf16(float f) {
  union { float f; unsigned int u; } c;
  c.f = f;
  unsigned int u = c.u;
  u += 0x7FFFu + ((u >> 16) & 1u);          // round-to-nearest-even
  return (unsigned short)(u >> 16);
}

DEV v8f vzero8() {
  v8f v;
#pragma unroll
  for (int i = 0; i < 8; ++i) v[i] = 0.0f;
  return v;
}

DEV v8f wmma_bf16(v16bf a, v16bf b, v8f c) {
  // D(16x16,f32) = A(16x32,bf16) * B(32x16,bf16) + C
  return __builtin_amdgcn_wmma_f32_16x16x32_bf16(false, a, false, b, (short)0, c,
                                                 false, false);
}

// LDS byte offset = low 32 bits of the flat pointer (ISA aperture mapping)
DEV unsigned int lds_off(const void* p) { return (unsigned int)(size_t)p; }

// async global -> LDS, 16 bytes per lane, tracked by ASYNCcnt
DEV void acopy_b128(unsigned int lds_byte_off, const unsigned short* g) {
  asm volatile("global_load_async_to_lds_b128 %0, %1, off"
               :: "v"(lds_byte_off), "v"(g)
               : "memory");
}

// ---------------- kernel 1a: x fp32 -> bf16 ----------------
__global__ __launch_bounds__(256) void xconv_kernel(const float* __restrict__ x,
                                                    unsigned short* __restrict__ xb) {
  int idx = blockIdx.x * 256 + threadIdx.x;            // over BT*Dc/4
  float4 f = reinterpret_cast<const float4*>(x)[idx];
  unsigned int lo = (unsigned int)f32_to_bf16(f.x) | ((unsigned int)f32_to_bf16(f.y) << 16);
  unsigned int hi = (unsigned int)f32_to_bf16(f.z) | ((unsigned int)f32_to_bf16(f.w) << 16);
  reinterpret_cast<uint2*>(xb)[idx] = make_uint2(lo, hi);
}

// ---------------- kernel 1b: W fp32 -> bf16 transposed (WT[n][k] = W[k][n]) -----
__global__ __launch_bounds__(256) void wtrans_kernel(
    const float* __restrict__ W0, const float* __restrict__ W1,
    const float* __restrict__ W2, const float* __restrict__ W3,
    unsigned short* __restrict__ T0, unsigned short* __restrict__ T1,
    unsigned short* __restrict__ T2, unsigned short* __restrict__ T3) {
  const float* W;
  unsigned short* T;
  switch (blockIdx.y) {
    case 0:  W = W0; T = T0; break;
    case 1:  W = W1; T = T1; break;
    case 2:  W = W2; T = T2; break;
    default: W = W3; T = T3; break;
  }
  int idx = blockIdx.x * 256 + threadIdx.x;            // over Dc*Dc
  int n = idx >> 10;
  int k = idx & (Dc - 1);
  T[idx] = f32_to_bf16(W[(size_t)k * Dc + n]);
}

// ---------------- shared GEMM mainloop, async double-buffered LDS -------------
// Block tile 128M x 128N, 8 waves (4x2), wave tile 32M x 64N, k-step 64.
// LDS buffers: sA0/sA1, sB0/sB1 each 128x64 bf16 (16 KB). Pipeline manually
// unrolled by 2 so every phase has compile-time-constant LDS addresses.
DEV void gemm_mainloop(const unsigned short* __restrict__ Aglob,
                       const unsigned short* __restrict__ Bglob,
                       unsigned short* sA0, unsigned short* sA1,
                       unsigned short* sB0, unsigned short* sB1,
                       int mblk, int nblk, int tid, v8f (&acc)[2][4]) {
  const int lane = tid & 31;
  const int wave = tid >> 5;
  const int half = lane >> 4;
  const int l16  = lane & 15;
  const int koff = half * 8;
  const int wm = wave >> 1, wn = wave & 1;

  // copy mapping: 1024 16B-chunks per 128x64 tile, 4 per thread
  const unsigned short* gA[4];
  const unsigned short* gB[4];
  int dst[4];
#pragma unroll
  for (int i = 0; i < 4; ++i) {
    const int c   = tid + i * 256;       // chunk 0..1023
    const int row = c >> 3;              // 0..127
    const int col = (c & 7) * 8;         // 0..56
    dst[i] = row * 64 + col;
    gA[i] = Aglob + (size_t)(mblk + row) * Dc + col;
    gB[i] = Bglob + (size_t)(nblk + row) * Dc + col;
  }

  auto issue = [&](unsigned short* dA, unsigned short* dB) {
#pragma unroll
    for (int i = 0; i < 4; ++i) {
      acopy_b128(lds_off(dA + dst[i]), gA[i]);
      acopy_b128(lds_off(dB + dst[i]), gB[i]);
      gA[i] += 64;
      gB[i] += 64;
    }
  };

  auto consume = [&](const unsigned short* bA, const unsigned short* bB) {
#pragma unroll
    for (int kk = 0; kk < 2; ++kk) {
      Frag a[2], bfr[4];
#pragma unroll
      for (int i = 0; i < 2; ++i) {
        const unsigned short* p = bA + (wm * 32 + i * 16 + l16) * 64 + kk * 32 + koff;
        a[i].q[0] = *reinterpret_cast<const uint4*>(p);
        a[i].q[1] = *reinterpret_cast<const uint4*>(p + 16);
      }
#pragma unroll
      for (int j = 0; j < 4; ++j) {
        const unsigned short* p = bB + (wn * 64 + j * 16 + l16) * 64 + kk * 32 + half * 8;
        bfr[j].q[0] = *reinterpret_cast<const uint4*>(p);
        bfr[j].q[1] = *reinterpret_cast<const uint4*>(p + 16);
      }
#pragma unroll
      for (int i = 0; i < 2; ++i)
#pragma unroll
        for (int j = 0; j < 4; ++j)
          acc[i][j] = wmma_bf16(a[i].v, bfr[j].v, acc[i][j]);
    }
  };

  constexpr int STEPS = Dc / 64;   // 16 (even)
  issue(sA0, sB0);
  for (int it = 0; it < STEPS; it += 2) {
    issue(sA1, sB1);
    asm volatile("s_wait_asynccnt 0x8" ::: "memory");
    __syncthreads();
    consume(sA0, sB0);
    __syncthreads();
    if (it + 2 < STEPS) {
      issue(sA0, sB0);
      asm volatile("s_wait_asynccnt 0x8" ::: "memory");
    } else {
      asm volatile("s_wait_asynccnt 0x0" ::: "memory");
    }
    __syncthreads();
    consume(sA1, sB1);
    __syncthreads();
  }
}

// ---------------- kernel 2: QKV projection GEMM ----------------
// grid (Dc/128, BT/128, 3), block 256
__global__ __launch_bounds__(256) void qkv_gemm_kernel(
    const unsigned short* __restrict__ xb,
    const unsigned short* __restrict__ WqT, const unsigned short* __restrict__ WkT,
    const unsigned short* __restrict__ WvT,
    const float* __restrict__ bq, const float* __restrict__ bk,
    const float* __restrict__ bv,
    unsigned short* __restrict__ Qb, unsigned short* __restrict__ Kb,
    unsigned short* __restrict__ Vt) {
  __shared__ unsigned short As[2][128 * 64];
  __shared__ unsigned short Bs[2][128 * 64];

  const int mode = blockIdx.z;
  const unsigned short* WT;
  const float* bias;
  if (mode == 0)      { WT = WqT; bias = bq; }
  else if (mode == 1) { WT = WkT; bias = bk; }
  else                { WT = WvT; bias = bv; }

  const int tid  = threadIdx.x;
  const int lane = tid & 31;
  const int wave = tid >> 5;
  const int half = lane >> 4;
  const int l16  = lane & 15;
  const int wm = wave >> 1, wn = wave & 1;
  const int m0 = blockIdx.y * 128 + wm * 32;
  const int n0 = blockIdx.x * 128 + wn * 64;

  v8f acc[2][4];
#pragma unroll
  for (int i = 0; i < 2; ++i)
#pragma unroll
    for (int j = 0; j < 4; ++j) acc[i][j] = vzero8();

  gemm_mainloop(xb, WT, &As[0][0], &As[1][0], &Bs[0][0], &Bs[1][0],
                blockIdx.y * 128, blockIdx.x * 128, tid, acc);

  // epilogue: +bias, convert bf16, scatter head-major (Q/K) or transposed (V)
#pragma unroll
  for (int i = 0; i < 2; ++i) {
#pragma unroll
    for (int j = 0; j < 4; ++j) {
      const int n  = n0 + j * 16 + l16;
      const float bb = bias[n];
      const int h  = n >> 7;
      const int dh = n & (DHc - 1);
      if (mode < 2) {
        unsigned short* out = (mode == 0) ? Qb : Kb;
#pragma unroll
        for (int e = 0; e < 8; ++e) {
          const int m = m0 + i * 16 + e + half * 8;
          const int bi = m >> 11;
          const int t  = m & (Tc - 1);
          out[((size_t)((bi * Hc + h) * Tc + t)) * DHc + dh] =
              f32_to_bf16(acc[i][j][e] + bb);
        }
      } else {
#pragma unroll
        for (int e = 0; e < 8; ++e) {
          const int m = m0 + i * 16 + e + half * 8;
          const int bi = m >> 11;
          const int t  = m & (Tc - 1);
          Vt[((size_t)((bi * Hc + h) * DHc + dh)) * Tc + t] =
              f32_to_bf16(acc[i][j][e] + bb);
        }
      }
    }
  }
}

// ---------------- kernel 3: flash attention with async K/V staging -----------
// grid (Tc/64, Bc*Hc), block 128 (4 waves, each owns 16 query rows; waves share
// the 32-key K/V tiles via double-buffered LDS, pipeline unrolled by 2)
__global__ __launch_bounds__(128) void attn_kernel(
    const unsigned short* __restrict__ Qb, const unsigned short* __restrict__ Kb,
    const unsigned short* __restrict__ Vt, unsigned short* __restrict__ ctx) {
  __shared__ unsigned short sK[2][32 * 128];   // [key][dh]
  __shared__ unsigned short sV[2][128 * 32];   // [dh][key]
  __shared__ unsigned short pbuf[4][16 * 32];  // per-wave P tile

  const int tid  = threadIdx.x;
  const int lane = tid & 31;
  const int wave = tid >> 5;
  const int half = lane >> 4;
  const int l16  = lane & 15;
  const int koff = half * 8;

  const int bh = blockIdx.y;
  const int b  = bh >> 3;
  const int h  = bh & 7;
  const int q0 = (blockIdx.x * 4 + wave) * 16;

  const unsigned short* Qh = Qb + (size_t)bh * Tc * DHc;
  const unsigned short* Kh = Kb + (size_t)bh * Tc * DHc;
  const unsigned short* Vh = Vt + (size_t)bh * DHc * Tc;

  // async copy mapping: 128 threads x 4 chunks cover each 8KB tile
  const unsigned short* gK[4];
  const unsigned short* gV[4];
  int kdst[4], vdst[4];
#pragma unroll
  for (int i = 0; i < 4; ++i) {
    const int c = tid + i * 128;                 // chunk id 0..511
    const int kr = c >> 4, kc = (c & 15) * 8;    // K tile: 32 x 128
    const int vr = c >> 2, vc = (c & 3) * 8;     // V tile: 128 x 32
    kdst[i] = kr * 128 + kc;
    vdst[i] = vr * 32 + vc;
    gK[i] = Kh + (size_t)kr * DHc + kc;          // advances 32*DHc per block
    gV[i] = Vh + (size_t)vr * Tc + vc;           // advances 32 per block
  }

  // Q fragments resident for the whole pass (DH=128 => 4 k-steps of 32)
  Frag qf[4];
#pragma unroll
  for (int kk = 0; kk < 4; ++kk) {
    const unsigned short* p = Qh + (size_t)(q0 + l16) * DHc + kk * 32 + koff;
    qf[kk].q[0] = *reinterpret_cast<const uint4*>(p);
    qf[kk].q[1] = *reinterpret_cast<const uint4*>(p + 16);
  }

  v8f o[8];
#pragma unroll
  for (int nt = 0; nt < 8; ++nt) o[nt] = vzero8();
  float mrow[8], lrow[8];
#pragma unroll
  for (int j = 0; j < 8; ++j) { mrow[j] = -1e30f; lrow[j] = 0.0f; }

  const float scale = 0.088388347648318447f;     // 1/sqrt(128)
  unsigned short* myp = pbuf[wave];

  auto issue = [&](unsigned short* dK, unsigned short* dV) {
#pragma unroll
    for (int i = 0; i < 4; ++i) {
      acopy_b128(lds_off(dK + kdst[i]), gK[i]);
      acopy_b128(lds_off(dV + vdst[i]), gV[i]);
      gK[i] += 32 * DHc;
      gV[i] += 32;
    }
  };

  auto consume = [&](const unsigned short* bK, const unsigned short* bV) {
    // ---- S = Q * K^T for 32 staged keys ----
    v8f s0 = vzero8(), s1 = vzero8();
#pragma unroll
    for (int kk = 0; kk < 4; ++kk) {
      Frag kba, kbb;
      const unsigned short* p0 = bK + (l16) * 128 + kk * 32 + half * 8;
      const unsigned short* p1 = bK + (16 + l16) * 128 + kk * 32 + half * 8;
      kba.q[0] = *reinterpret_cast<const uint4*>(p0);
      kba.q[1] = *reinterpret_cast<const uint4*>(p0 + 16);
      kbb.q[0] = *reinterpret_cast<const uint4*>(p1);
      kbb.q[1] = *reinterpret_cast<const uint4*>(p1 + 16);
      s0 = wmma_bf16(qf[kk].v, kba.v, s0);
      s1 = wmma_bf16(qf[kk].v, kbb.v, s1);
    }

    // ---- online softmax (row = acc element + 8*half, col = lane&15) ----
    float alpha[8];
#pragma unroll
    for (int j = 0; j < 8; ++j) {
      float a0 = s0[j] * scale;
      float a1 = s1[j] * scale;
      float mx = fmaxf(a0, a1);
#pragma unroll
      for (int off = 8; off; off >>= 1) mx = fmaxf(mx, __shfl_xor(mx, off, 32));
      const float mnew = fmaxf(mrow[j], mx);
      alpha[j] = __expf(mrow[j] - mnew);
      const float p0 = __expf(a0 - mnew);
      const float p1 = __expf(a1 - mnew);
      float rs = p0 + p1;
#pragma unroll
      for (int off = 8; off; off >>= 1) rs += __shfl_xor(rs, off, 32);
      lrow[j] = lrow[j] * alpha[j] + rs;
      mrow[j] = mnew;
      const int r = j + half * 8;
      myp[r * 32 + l16]      = f32_to_bf16(p0);
      myp[r * 32 + 16 + l16] = f32_to_bf16(p1);
    }
#pragma unroll
    for (int nt = 0; nt < 8; ++nt)
#pragma unroll
      for (int j = 0; j < 8; ++j) o[nt][j] = o[nt][j] * alpha[j];

    // P tile: C-layout -> A-layout via per-wave LDS round trip
    asm volatile("s_wait_dscnt 0x0" ::: "memory");
    Frag pf;
    {
      const unsigned short* pp = myp + l16 * 32 + koff;
      pf.q[0] = *reinterpret_cast<const uint4*>(pp);
      pf.q[1] = *reinterpret_cast<const uint4*>(pp + 16);
    }

    // ---- O += P * V from staged V (rows = dh, contiguous keys) ----
#pragma unroll
    for (int nt = 0; nt < 8; ++nt) {
      Frag vf;
      const unsigned short* pv = bV + (nt * 16 + l16) * 32 + half * 8;
      vf.q[0] = *reinterpret_cast<const uint4*>(pv);
      vf.q[1] = *reinterpret_cast<const uint4*>(pv + 16);
      o[nt] = wmma_bf16(pf.v, vf.v, o[nt]);
    }
  };

  constexpr int STEPS = Tc / 32;   // 64 (even)
  issue(&sK[0][0], &sV[0][0]);
  for (int it = 0; it < STEPS; it += 2) {
    issue(&sK[1][0], &sV[1][0]);
    asm volatile("s_wait_asynccnt 0x8" ::: "memory");
    __syncthreads();
    consume(&sK[0][0], &sV[0][0]);
    __syncthreads();
    if (it + 2 < STEPS) {
      issue(&sK[0][0], &sV[0][0]);
      asm volatile("s_wait_asynccnt 0x8" ::: "memory");
    } else {
      asm volatile("s_wait_asynccnt 0x0" ::: "memory");
    }
    __syncthreads();
    consume(&sK[1][0], &sV[1][0]);
    __syncthreads();
  }

  // ---- normalize and write ctx back to [B,T,D] (bf16) ----
#pragma unroll
  for (int j = 0; j < 8; ++j) {
    const float inv = 1.0f / lrow[j];
    const int t = q0 + j + half * 8;
    const size_t rowbase = ((size_t)(b * Tc + t)) * Dc + h * DHc;
#pragma unroll
    for (int nt = 0; nt < 8; ++nt)
      ctx[rowbase + nt * 16 + l16] = f32_to_bf16(o[nt][j] * inv);
  }
}

// ---------------- kernel 4: out-proj GEMM + bias + residual ----------------
// grid (Dc/128, BT/128), block 256
__global__ __launch_bounds__(256) void oproj_kernel(
    const unsigned short* __restrict__ ctxb, const unsigned short* __restrict__ WoT,
    const float* __restrict__ bo, const float* __restrict__ x,
    float* __restrict__ y) {
  __shared__ unsigned short As[2][128 * 64];
  __shared__ unsigned short Bs[2][128 * 64];

  const int tid  = threadIdx.x;
  const int lane = tid & 31;
  const int wave = tid >> 5;
  const int half = lane >> 4;
  const int l16  = lane & 15;
  const int wm = wave >> 1, wn = wave & 1;
  const int m0 = blockIdx.y * 128 + wm * 32;
  const int n0 = blockIdx.x * 128 + wn * 64;

  v8f acc[2][4];
#pragma unroll
  for (int i = 0; i < 2; ++i)
#pragma unroll
    for (int j = 0; j < 4; ++j) acc[i][j] = vzero8();

  gemm_mainloop(ctxb, WoT, &As[0][0], &As[1][0], &Bs[0][0], &Bs[1][0],
                blockIdx.y * 128, blockIdx.x * 128, tid, acc);

#pragma unroll
  for (int i = 0; i < 2; ++i) {
#pragma unroll
    for (int j = 0; j < 4; ++j) {
      const int n = n0 + j * 16 + l16;
      const float bb = bo[n];
#pragma unroll
      for (int e = 0; e < 8; ++e) {
        const int m = m0 + i * 16 + e + half * 8;
        const size_t idx = (size_t)m * Dc + n;
        y[idx] = acc[i][j][e] + bb + x[idx];
      }
    }
  }
}

// ---------------- kernel 5: in-place LayerNorm, one block per row -------------
__global__ __launch_bounds__(256) void ln_kernel(float* __restrict__ y,
                                                 const float* __restrict__ gamma,
                                                 const float* __restrict__ beta) {
  const int row = blockIdx.x;
  float* p = y + (size_t)row * Dc;
  float v[4];
  float s = 0.0f, ss = 0.0f;
#pragma unroll
  for (int i = 0; i < 4; ++i) {
    const float t = p[threadIdx.x + i * 256];
    v[i] = t; s += t; ss += t * t;
  }
#pragma unroll
  for (int off = 16; off; off >>= 1) {
    s  += __shfl_xor(s, off, 32);
    ss += __shfl_xor(ss, off, 32);
  }
  __shared__ float shs[8], shss[8];
  if ((threadIdx.x & 31) == 0) {
    shs[threadIdx.x >> 5]  = s;
    shss[threadIdx.x >> 5] = ss;
  }
  __syncthreads();
  s = 0.0f; ss = 0.0f;
#pragma unroll
  for (int i = 0; i < 8; ++i) { s += shs[i]; ss += shss[i]; }
  const float mu   = s * (1.0f / (float)Dc);
  const float var  = ss * (1.0f / (float)Dc) - mu * mu;
  const float rstd = rsqrtf(var + 1e-5f);
#pragma unroll
  for (int i = 0; i < 4; ++i) {
    const int c = threadIdx.x + i * 256;
    p[c] = (v[i] - mu) * rstd * gamma[c] + beta[c];
  }
}

// ---------------- host launch ----------------
extern "C" void kernel_launch(void* const* d_in, const int* in_sizes, int n_in,
                              void* d_out, int out_size, void* d_ws, size_t ws_size,
                              hipStream_t stream) {
  (void)in_sizes; (void)n_in; (void)out_size; (void)ws_size;
  const float* x     = (const float*)d_in[0];
  const float* Wq    = (const float*)d_in[1];
  const float* bq    = (const float*)d_in[2];
  const float* Wk    = (const float*)d_in[3];
  const float* bk    = (const float*)d_in[4];
  const float* Wv    = (const float*)d_in[5];
  const float* bv    = (const float*)d_in[6];
  const float* Wo    = (const float*)d_in[7];
  const float* bo    = (const float*)d_in[8];
  const float* gamma = (const float*)d_in[9];
  const float* beta  = (const float*)d_in[10];
  float* out = (float*)d_out;

  char* ws = (char*)d_ws;
  const size_t SZ_XB   = (size_t)BT * Dc * 2;             // 16 MB
  const size_t SZ_W    = (size_t)Dc * Dc * 2;             //  2 MB each
  const size_t SZ_HEAD = (size_t)Bc * Hc * Tc * DHc * 2;  // 16 MB each
  unsigned short* xb  = (unsigned short*)(ws);
  unsigned short* WqT = (unsigned short*)(ws + SZ_XB);
  unsigned short* WkT = (unsigned short*)(ws + SZ_XB + SZ_W);
  unsigned short* WvT = (unsigned short*)(ws + SZ_XB + 2 * SZ_W);
  unsigned short* WoT = (unsigned short*)(ws + SZ_XB + 3 * SZ_W);
  unsigned short* Qb  = (unsigned short*)(ws + SZ_XB + 4 * SZ_W);
  unsigned short* Kb  = (unsigned short*)(ws + SZ_XB + 4 * SZ_W + SZ_HEAD);
  unsigned short* Vt  = (unsigned short*)(ws + SZ_XB + 4 * SZ_W + 2 * SZ_HEAD);
  unsigned short* Cb  = (unsigned short*)(ws + SZ_XB + 4 * SZ_W + 3 * SZ_HEAD);

  xconv_kernel<<<(BT * Dc / 4) / 256, 256, 0, stream>>>(x, xb);
  wtrans_kernel<<<dim3(Dc * Dc / 256, 4), 256, 0, stream>>>(Wq, Wk, Wv, Wo,
                                                            WqT, WkT, WvT, WoT);
  qkv_gemm_kernel<<<dim3(Dc / 128, BT / 128, 3), 256, 0, stream>>>(
      xb, WqT, WkT, WvT, bq, bk, bv, Qb, Kb, Vt);
  attn_kernel<<<dim3(Tc / 64, Bc * Hc), 128, 0, stream>>>(Qb, Kb, Vt, Cb);
  oproj_kernel<<<dim3(Dc / 128, BT / 128), 256, 0, stream>>>(Cb, WoT, bo, x, out);
  ln_kernel<<<BT, 256, 0, stream>>>(out, gamma, beta);
}
